// GCN_33998961115490
// MI455X (gfx1250) — compile-verified, matrix operand order checked by reading, not run on hardware
//
#include <hip/hip_runtime.h>
#include <hip/hip_bf16.h>
#include <math.h>

typedef __attribute__((ext_vector_type(16))) _Float16 v16h;
typedef __attribute__((ext_vector_type(8)))  float    v8f;
typedef int b128_t __attribute__((vector_size(16)));   /* matches builtin param type */

#define N_NODES   50000
#define N_EDGES   800000
#define N_MSGS    (N_EDGES + N_NODES)   /* edges + self loops */
#define NUM_GRAPHS 64
#define NFEAT     128                   /* NIN == NHID == 128 */
#define NOUT      10
#define BFRAG_HALVES (4 * 8 * 32 * 16)  /* packed weight: 16384 halves = 32 KB */

/* ---------------- degree / norm ---------------- */

__global__ void k_init_deg(float* deg) {
    int i = blockIdx.x * blockDim.x + threadIdx.x;
    if (i < N_NODES) deg[i] = 1.0f;     /* self loop contributes 1 */
}

__global__ void k_deg_accum(const int* __restrict__ dst, float* __restrict__ deg) {
    int e = blockIdx.x * blockDim.x + threadIdx.x;
    if (e < N_EDGES) atomicAdd(&deg[dst[e]], 1.0f);
}

__global__ void k_dinv(const float* __restrict__ deg, float* __restrict__ dinv) {
    int i = blockIdx.x * blockDim.x + threadIdx.x;
    if (i < N_NODES) dinv[i] = rsqrtf(deg[i]);   /* deg >= 1 always */
}

/* ---------------- weight packing into WMMA B-fragment layout ----------------
 * Packed index t = ((ks*8 + nt)*32 + lane)*16 + i
 * lane holds column n = nt*16 + (lane&15); K pattern mirrors the A fragment:
 * kbase = (lane<16)?0:8 ; i<8 -> k = ks*32 + kbase + i ; i>=8 -> k = ks*32 + kbase + 8 + i
 */
__global__ void k_pack_w(const float* __restrict__ W, _Float16* __restrict__ P) {
    int t = blockIdx.x * blockDim.x + threadIdx.x;
    if (t >= BFRAG_HALVES) return;
    int i    =  t        & 15;
    int lane = (t >> 4)  & 31;
    int nt   = (t >> 9)  & 7;
    int ks   =  t >> 12;
    int kbase = (lane < 16) ? 0 : 8;
    int k = ks * 32 + ((i < 8) ? (kbase + i) : (kbase + 8 + i));
    int n = nt * 16 + (lane & 15);
    P[t] = (_Float16)W[k * NFEAT + n];
}

/* ---------------- WMMA GEMM: C[M,128] = op(A)[M,128] @ W[128,128] ----------------
 * One wave per 16-row M tile; 8 N-tiles x 4 K-steps of v_wmma_f32_16x16x32_f16.
 * 8 waves (256 threads) per block. M = 50000 = 3125 * 16 exactly.
 * The full packed weight (32 KB) is staged into LDS once per block via the
 * gfx1250 async global->LDS path (ASYNCcnt), then WMMA feeds from ds_load.
 */
template<bool RELU_IN>
__global__ void k_gemm_wmma(const float* __restrict__ A,
                            const _Float16* __restrict__ Bp,
                            float* __restrict__ C, int M) {
    __shared__ __attribute__((aligned(32))) _Float16 Bs[BFRAG_HALVES];

    /* cooperative staging: 256 threads x 8 chunks x 16 B = 32 KB */
#if __has_builtin(__builtin_amdgcn_global_load_async_to_lds_b128)
#pragma unroll
    for (int c = 0; c < 8; ++c) {
        int boff = (c * 256 + (int)threadIdx.x) * 16;   /* byte offset */
        __builtin_amdgcn_global_load_async_to_lds_b128(
            (__attribute__((address_space(1))) b128_t*)((const char*)Bp + boff),
            (__attribute__((address_space(3))) b128_t*)((char*)Bs + boff),
            0, 0);
    }
#if __has_builtin(__builtin_amdgcn_s_wait_asynccnt)
    __builtin_amdgcn_s_wait_asynccnt(0);
#else
    asm volatile("s_wait_asynccnt 0" ::: "memory");
#endif
#else
#pragma unroll
    for (int c = 0; c < 8; ++c) {
        int boff = (c * 256 + (int)threadIdx.x) * 16;
        *(float4*)((char*)Bs + boff) = *(const float4*)((const char*)Bp + boff);
    }
#endif
    __syncthreads();

    const int wave = threadIdx.x >> 5;
    const int lane = threadIdx.x & 31;
    const int mtile = blockIdx.x * 8 + wave;
    if (mtile * 16 >= M) return;                 /* uniform per wave: EXEC stays full */

    const int row   = mtile * 16 + (lane & 15);
    const int kbase = (lane < 16) ? 0 : 8;
    const float* arow = A + (size_t)row * NFEAT;

    v16h a[4];
#pragma unroll
    for (int ks = 0; ks < 4; ++ks) {
#pragma unroll
        for (int i = 0; i < 8; ++i) {
            float f0 = arow[ks * 32 + kbase + i];
            float f1 = arow[ks * 32 + kbase + 16 + i];
            if (RELU_IN) { f0 = fmaxf(f0, 0.0f); f1 = fmaxf(f1, 0.0f); }
            a[ks][i]     = (_Float16)f0;
            a[ks][i + 8] = (_Float16)f1;
        }
    }

    const int r0  = mtile * 16 + ((lane < 16) ? 0 : 8);
    const int nlo = lane & 15;
#pragma unroll
    for (int nt = 0; nt < 8; ++nt) {
        v8f acc = {};
#pragma unroll
        for (int ks = 0; ks < 4; ++ks) {
            v16h b = *(const v16h*)(Bs + (size_t)(((ks * 8 + nt) * 32 + lane) * 16));
            acc = __builtin_amdgcn_wmma_f32_16x16x32_f16(
                      false, a[ks], false, b, (short)0, acc, false, false);
        }
        float* cptr = C + (size_t)r0 * NFEAT + nt * 16 + nlo;
#pragma unroll
        for (int r = 0; r < 8; ++r) cptr[(size_t)r * NFEAT] = acc[r];
    }
}

/* ---------------- accumulator init with bias ---------------- */
__global__ void k_init_bias(float* __restrict__ AGG, const float* __restrict__ bias) {
    long long g = (long long)blockIdx.x * blockDim.x + threadIdx.x;
    if (g < (long long)N_NODES * NFEAT) AGG[g] = bias[(int)(g & (NFEAT - 1))];
}

/* ---------------- message pass: one wave per message, lane = float4 ---------------- */
__global__ void k_scatter(const int* __restrict__ src, const int* __restrict__ dst,
                          const float* __restrict__ dinv,
                          const float* __restrict__ H, float* __restrict__ AGG) {
    long long gid = (long long)blockIdx.x * blockDim.x + threadIdx.x;
    int msg  = (int)(gid >> 5);
    int lane = threadIdx.x & 31;
    if (msg >= N_MSGS) return;
    int s, d;
    if (msg < N_EDGES) { s = src[msg]; d = dst[msg]; }
    else               { s = d = msg - N_EDGES; }
    float w = dinv[s] * dinv[d];
    float4 hv = *(const float4*)(H + (size_t)s * NFEAT + lane * 4);
    float* out = AGG + (size_t)d * NFEAT + lane * 4;
    __builtin_prefetch(out, 1, 1);               /* gfx1250 global_prefetch path */
    atomicAdd(out + 0, hv.x * w);
    atomicAdd(out + 1, hv.y * w);
    atomicAdd(out + 2, hv.z * w);
    atomicAdd(out + 3, hv.w * w);
}

/* ---------------- pooled init ---------------- */
__global__ void k_init_pooled(float* pooled) {
    int i = blockIdx.x * blockDim.x + threadIdx.x;
    if (i < NUM_GRAPHS * NFEAT) pooled[i] = 0.0f;
}

/* ---------------- per-node max (y) + segment-max pooling ----------------
 * h = relu(AGG2) >= 0, so uint-bit atomicMax against 0-initialized pooled is exact.
 */
__global__ void k_pool(const float* __restrict__ AGG2, const int* __restrict__ batch,
                       float* __restrict__ y, float* __restrict__ pooled) {
    int node = blockIdx.x * (blockDim.x >> 5) + (threadIdx.x >> 5);
    int lane = threadIdx.x & 31;
    if (node >= N_NODES) return;
    float4 v = *(const float4*)(AGG2 + (size_t)node * NFEAT + lane * 4);
    v.x = fmaxf(v.x, 0.0f); v.y = fmaxf(v.y, 0.0f);
    v.z = fmaxf(v.z, 0.0f); v.w = fmaxf(v.w, 0.0f);
    float m = fmaxf(fmaxf(v.x, v.y), fmaxf(v.z, v.w));
#pragma unroll
    for (int off = 16; off >= 1; off >>= 1) m = fmaxf(m, __shfl_xor(m, off, 32));
    if (lane == 0) y[node] = m;
    int g = batch[node];
    unsigned* pb = (unsigned*)(pooled + (size_t)g * NFEAT + lane * 4);
    atomicMax(pb + 0, __float_as_uint(v.x));
    atomicMax(pb + 1, __float_as_uint(v.y));
    atomicMax(pb + 2, __float_as_uint(v.z));
    atomicMax(pb + 3, __float_as_uint(v.w));
}

/* ---------------- head: logits = pooled @ Wd + bd, softmax ---------------- */
__global__ void k_head(const float* __restrict__ pooled, const float* __restrict__ Wd,
                       const float* __restrict__ bd, float* __restrict__ probs) {
    int g = blockIdx.x;
    int lane = threadIdx.x;                      /* blockDim = 32 */
    float logit = -INFINITY;
    if (lane < NOUT) {
        float acc = bd[lane];
        const float* p = pooled + (size_t)g * NFEAT;
#pragma unroll 4
        for (int k = 0; k < NFEAT; ++k) acc += p[k] * Wd[k * NOUT + lane];
        logit = acc;
    }
    float m = logit;
#pragma unroll
    for (int off = 16; off >= 1; off >>= 1) m = fmaxf(m, __shfl_xor(m, off, 32));
    float e = (lane < NOUT) ? expf(logit - m) : 0.0f;
    float s = e;
#pragma unroll
    for (int off = 16; off >= 1; off >>= 1) s += __shfl_xor(s, off, 32);
    if (lane < NOUT) probs[g * NOUT + lane] = e / s;
}

/* ---------------- host launcher ---------------- */
extern "C" void kernel_launch(void* const* d_in, const int* in_sizes, int n_in,
                              void* d_out, int out_size, void* d_ws, size_t ws_size,
                              hipStream_t stream) {
    const float* x    = (const float*)d_in[0];
    const int*   ei   = (const int*)  d_in[1];     /* [2, E] flat */
    const int*   batch= (const int*)  d_in[2];
    const float* W1   = (const float*)d_in[3];
    const float* b1   = (const float*)d_in[4];
    const float* W2   = (const float*)d_in[5];
    const float* b2   = (const float*)d_in[6];
    const float* Wd   = (const float*)d_in[7];
    const float* bd   = (const float*)d_in[8];
    const int* src = ei;
    const int* dst = ei + N_EDGES;

    float* probs = (float*)d_out;                     /* [64,10] */
    float* y     = (float*)d_out + NUM_GRAPHS * NOUT; /* [N,1]   */

    /* workspace carve-up (256B aligned) */
    char* ws = (char*)d_ws;
    size_t off = 0;
    auto carve = [&](size_t bytes) { size_t o = off; off += (bytes + 255) & ~(size_t)255; return o; };
    float*    dinv   = (float*)   (ws + carve((size_t)N_NODES * 4));
    _Float16* W1p    = (_Float16*)(ws + carve((size_t)BFRAG_HALVES * 2));
    _Float16* W2p    = (_Float16*)(ws + carve((size_t)BFRAG_HALVES * 2));
    float*    pooled = (float*)   (ws + carve((size_t)NUM_GRAPHS * NFEAT * 4));
    float*    bufA   = (float*)   (ws + carve((size_t)N_NODES * NFEAT * 4));
    float*    bufB   = (float*)   (ws + carve((size_t)N_NODES * NFEAT * 4));

    const int B = 256;
    const int gN      = (N_NODES + B - 1) / B;
    const int gE      = (N_EDGES + B - 1) / B;
    const int gPack   = BFRAG_HALVES / B;
    const int gGemm   = (N_NODES / 16 + 7) / 8;               /* 3125 mtiles / 8 waves */
    const int gFeat   = (int)(((long long)N_NODES * NFEAT + B - 1) / B);
    const int gMsg    = (int)(((long long)N_MSGS * 32 + B - 1) / B);
    const int gPool   = N_NODES / 8;                          /* 8 waves/block, exact */

    /* norm: deg (in bufA scratch) -> dinv */
    k_init_deg <<<gN, B, 0, stream>>>(bufA);
    k_deg_accum<<<gE, B, 0, stream>>>(dst, bufA);
    k_dinv     <<<gN, B, 0, stream>>>(bufA, dinv);

    /* pack weights into WMMA B-fragment layout */
    k_pack_w<<<gPack, B, 0, stream>>>(W1, W1p);
    k_pack_w<<<gPack, B, 0, stream>>>(W2, W2p);

    /* conv1: h1 = x @ W1 ; agg1 = scatter(norm * h1[src]) + b1 */
    k_gemm_wmma<false><<<gGemm, B, 0, stream>>>(x, W1p, bufA, N_NODES);
    k_init_bias<<<gFeat, B, 0, stream>>>(bufB, b1);
    k_scatter  <<<gMsg, B, 0, stream>>>(src, dst, dinv, bufA, bufB);

    /* conv2: h2 = relu(agg1) @ W2 ; agg2 = scatter(...) + b2 */
    k_gemm_wmma<true><<<gGemm, B, 0, stream>>>(bufB, W2p, bufA, N_NODES);
    k_init_bias<<<gFeat, B, 0, stream>>>(bufB, b2);
    k_scatter  <<<gMsg, B, 0, stream>>>(src, dst, dinv, bufA, bufB);

    /* pooling: y = per-node max of relu(agg2); pooled = segment_max over batch */
    k_init_pooled<<<(NUM_GRAPHS * NFEAT + B - 1) / B, B, 0, stream>>>(pooled);
    k_pool<<<gPool, B, 0, stream>>>(bufB, batch, y, pooled);

    /* head: softmax(pooled @ Wd + bd) */
    k_head<<<NUM_GRAPHS, 32, 0, stream>>>(pooled, Wd, bd, probs);

    (void)in_sizes; (void)n_in; (void)out_size; (void)ws_size;
}
